// AtomTypeGNN_4810363372605
// MI455X (gfx1250) — matrix operand order repaired and checked by reading, not compiled, not run on hardware
//
#include <hip/hip_runtime.h>
#include <hip/hip_bf16.h>
#include <math.h>

// Problem sizes (fixed by the reference)
#define NA   2048   // atoms (m == n)
#define KD   64     // dist_exp_size (f)
#define HD   128    // atom_emb_size (h)
#define OUTD 128    // hidden_size (k)

typedef __attribute__((ext_vector_type(16))) __bf16 v16bf;
typedef __attribute__((ext_vector_type(8)))  float  v8f;
typedef __attribute__((ext_vector_type(4)))  float  v4f;

// ---------------------------------------------------------------------------
// Kernel 1: adj_exp[m,k] = sum_n dist_adj[m,n] * dist_exp[m,n,k]
// Bandwidth-bound: streams 1 GiB of dist_exp once. One block per row m,
// 256 threads = 16 n-slices x 16 k-quads (v4f), LDS reduction.
// ---------------------------------------------------------------------------
__global__ void k_adj_exp(const float* __restrict__ dist_adj,
                          const float* __restrict__ dist_exp,
                          float* __restrict__ adj_exp) {
    const int m  = blockIdx.x;
    const int t  = threadIdx.x;     // 0..255
    const int kq = t & 15;          // k quad: covers k = kq*4 .. kq*4+3
    const int ns = t >> 4;          // n slice 0..15

    const float* __restrict__ adj_row = dist_adj + (size_t)m * NA;
    const v4f*   __restrict__ de = (const v4f*)(dist_exp + (size_t)m * NA * KD);

    v4f acc = {0.0f, 0.0f, 0.0f, 0.0f};
    for (int n = ns; n < NA; n += 16) {
        const float a = adj_row[n];
        // stream-once data: non-temporal 128-bit load
        v4f v = __builtin_nontemporal_load(&de[(size_t)n * (KD / 4) + kq]);
        acc += a * v;
    }

    __shared__ v4f red[256];
    red[t] = acc;
    __syncthreads();
    if (ns == 0) {
        v4f s = red[kq];
        #pragma unroll
        for (int j = 1; j < 16; ++j)
            s += red[kq + (j << 4)];
        ((v4f*)(adj_exp + (size_t)m * KD))[kq] = s;
    }
}

// ---------------------------------------------------------------------------
// Kernel 0: pre-swizzle bilinear_w (f32 [8192,128]) into bf16 B-fragments.
// K-depth index d = f*128 + h, step s = d/32 = f*4 + hs.  For step s,
// col-tile t, lane l, element e:
//   wp[(((s*8+t)*32 + l)*16) + e] = W[s*32 + (l>>4)*16 + e, t*16 + (l&15)]
// Each lane's 16 bf16 are contiguous (32 B) -> coalesced b128 loads.
// ---------------------------------------------------------------------------
__global__ void k_pack_w(const float* __restrict__ w, __bf16* __restrict__ wp) {
    const int g    = blockIdx.x * blockDim.x + threadIdx.x;  // 0..65535
    const int lane = g & 31;
    const int t    = (g >> 5) & 7;
    const int s    = g >> 8;
    const int drow = s * 32 + (lane >> 4) * 16;
    const int col  = t * 16 + (lane & 15);

    v16bf v;
    #pragma unroll
    for (int e = 0; e < 16; ++e)
        v[e] = (__bf16)w[(size_t)(drow + e) * OUTD + col];

    *(v16bf*)(wp + ((size_t)((s * 8 + t) * 32 + lane) << 4)) = v;
}

// ---------------------------------------------------------------------------
// Kernel 2: fused bilinear + softplus + bias via WMMA bf16.
//   feat[n,k] = sum_f adj_exp[n,f] * ( sum_h emb[n,h] * W[f,h,k] )
// Block = 256 threads (8 waves); block owns a 16-row strip of n; wave w owns
// the 16-col output tile w.  The emb A-fragments (4 x K32 covering h=128)
// are built ONCE and reused for all 64 f: per f the wave runs 4 chained
// WMMAs into g (f32), then folds adjT[f][row]*g into the f32 output acc
// with two aligned v4f LDS broadcasts (no register transposes).
// ---------------------------------------------------------------------------
#define ADJT_STRIDE 20   // floats per row: 16 + 4 pad -> 80 B, 16 B aligned

__global__ void k_bilinear_wmma(const float* __restrict__ adj_exp,
                                const float* __restrict__ atom_emb,
                                const __bf16* __restrict__ wp,
                                const float* __restrict__ bias,
                                float* __restrict__ out) {
    __shared__ float s_adjT[KD][ADJT_STRIDE];   // transposed: [f][row]
    __shared__ float s_emb[16][HD + 4];

    const int n0  = blockIdx.x * 16;
    const int tid = threadIdx.x;

    // adj: global-coalesced read (consecutive f), transposed LDS write
    for (int i = tid; i < 16 * KD; i += 256) {
        const int r = i >> 6, f = i & 63;
        s_adjT[f][r] = adj_exp[(size_t)(n0 + r) * KD + f];
    }
    for (int i = tid; i < 16 * HD; i += 256) {
        const int r = i >> 7, c = i & 127;
        s_emb[r][c] = atom_emb[(size_t)(n0 + r) * HD + c];
    }
    __syncthreads();

    const int lane  = tid & 31;
    const int wave  = tid >> 5;        // output col tile 0..7
    const int mrow  = lane & 15;       // A-matrix M row held by this lane
    const int khalf = lane >> 4;       // K half-interleave
    const int kb0   = khalf * 8;

    // Pre-build the 4 emb A-fragments (16-bit A 16x32 layout, ISA 7.12.2):
    // elems 0..15 -> K = khalf*16 + e (contiguous, matches packed B)
    v16bf Af[4];
    #pragma unroll
    for (int hs = 0; hs < 4; ++hs) {
        const int hb = hs << 5;
        #pragma unroll
        for (int j = 0; j < 8; ++j) {
            Af[hs][j]     = (__bf16)s_emb[mrow][hb + 2 * kb0 + j];
            Af[hs][j + 8] = (__bf16)s_emb[mrow][hb + 2 * kb0 + 8 + j];
        }
    }

    const __bf16* __restrict__ wpw =
        wp + ((size_t)(wave * 32 + lane) << 4);     // + s*4096 per K-step

    const int rbase = khalf * 8;                    // C/D row base for this lane
    v4f accL = {0.f, 0.f, 0.f, 0.f};                // acc for r = 0..3
    v4f accH = {0.f, 0.f, 0.f, 0.f};                // acc for r = 4..7

    for (int f = 0; f < KD; ++f) {
        v8f g = {};
        #pragma unroll
        for (int hs = 0; hs < 4; ++hs) {
            const v16bf B =
                *(const v16bf*)(wpw + ((size_t)(f * 4 + hs) << 12));
            g = __builtin_amdgcn_wmma_f32_16x16x32_bf16(
                    false, Af[hs], false, B, (short)0, g, false, false);
        }
        // two aligned 128-bit LDS broadcasts of adjT[f][rbase..rbase+7]
        const v4f aL = *(const v4f*)&s_adjT[f][rbase];
        const v4f aH = *(const v4f*)&s_adjT[f][rbase + 4];
        v4f gL = {g[0], g[1], g[2], g[3]};
        v4f gH = {g[4], g[5], g[6], g[7]};
        accL += aL * gL;
        accH += aH * gH;
    }

    // C/D layout: col = lane&15 ; VGPR r -> row = khalf*8 + r
    const int   col = wave * 16 + (lane & 15);
    const float b   = bias[col];
    float accf[8] = {accL.x, accL.y, accL.z, accL.w,
                     accH.x, accH.y, accH.z, accH.w};
    #pragma unroll
    for (int r = 0; r < 8; ++r) {
        const int   row = n0 + rbase + r;
        const float x   = accf[r];
        // softplus via HW transcendentals (v_exp_f32 / v_log_f32)
        const float sp  = (x > 20.0f) ? x : __logf(1.0f + __expf(x));
        out[(size_t)row * OUTD + col] = sp + b;
    }
}

// ---------------------------------------------------------------------------
extern "C" void kernel_launch(void* const* d_in, const int* in_sizes, int n_in,
                              void* d_out, int out_size, void* d_ws, size_t ws_size,
                              hipStream_t stream) {
    const float* dist_adj   = (const float*)d_in[0];  // [2048,2048]
    const float* dist_exp   = (const float*)d_in[1];  // [2048,2048,64]
    const float* atom_emb   = (const float*)d_in[2];  // [2048,128]
    const float* bilinear_w = (const float*)d_in[3];  // [64,128,128]
    const float* bilinear_b = (const float*)d_in[4];  // [128]
    float* out = (float*)d_out;                       // [2048,128] f32

    // workspace: [0, 512KB) adj_exp f32 ; [512KB, 512KB+2MB) packed W bf16
    float*  adj_exp = (float*)d_ws;
    __bf16* wp      = (__bf16*)((char*)d_ws + (size_t)NA * KD * sizeof(float));

    k_pack_w<<<256, 256, 0, stream>>>(bilinear_w, wp);
    k_adj_exp<<<NA, 256, 0, stream>>>(dist_adj, dist_exp, adj_exp);
    k_bilinear_wmma<<<NA / 16, 256, 0, stream>>>(adj_exp, atom_emb, wp,
                                                 bilinear_b, out);
}
